// DisentangledSelfAttention_18769007084654
// MI455X (gfx1250) — matrix operand order repaired and checked by reading, not compiled, not run on hardware
//
#include <hip/hip_runtime.h>
#include <math.h>

// ---------------------------------------------------------------------------
// DeBERTa-style disentangled self-attention for MI455X (gfx1250, wave32).
// All matmuls run through v_wmma_f32_16x16x32_bf16 (fp32 accumulate).
// GEMM: 128x64 block tile, 8 waves, 32x32 per-wave tile -> 4 WMMAs / K-step
// with A/B fragment reuse. Attention phase serialized over batch so per-batch
// scratch (~59MB) stays resident in the 192MB L2.
// ---------------------------------------------------------------------------

#define NB   8
#define SEQ  512
#define NH   16
#define HID  1024
#define HD   64
#define MAXP 512
#define BUCK 256

#define LDP  40   // padded LDS row stride (bf16 elems): 80B, 16B-aligned

typedef __attribute__((ext_vector_type(16))) __bf16 v16bf;
typedef __attribute__((ext_vector_type(8)))  float  v8f;

static constexpr float SCALE_F = 0.07216878364870322f; // 1/sqrt(3*64)

// ---------------- elementwise f32 -> bf16 ----------------
__global__ __launch_bounds__(256)
void cvt_f32_bf16(const float* __restrict__ src, __bf16* __restrict__ dst, int n) {
  int i = blockIdx.x * 256 + threadIdx.x;
  if (i < n) dst[i] = (__bf16)src[i];
}

// ---------------- 1024x1024 transpose + convert (weights -> W^T bf16) -------
__global__ __launch_bounds__(256)
void transpose_cvt(const float* __restrict__ src, __bf16* __restrict__ dst) {
  __shared__ float tile[16][17];
  int x = blockIdx.x * 16 + threadIdx.x;
  int y = blockIdx.y * 16 + threadIdx.y;
  tile[threadIdx.y][threadIdx.x] = src[(long)y * HID + x];
  __syncthreads();
  int ox = blockIdx.y * 16 + threadIdx.x;
  int oy = blockIdx.x * 16 + threadIdx.y;
  dst[(long)oy * HID + ox] = (__bf16)tile[threadIdx.x][threadIdx.y];
}

// ---------------- log-bucket relative-position index tables -----------------
// index depends only on delta = q - k in [-511, 511]
__global__ void build_pos_tables(int* __restrict__ c2ptab, int* __restrict__ p2ctab) {
  int i = blockIdx.x * blockDim.x + threadIdx.x;
  if (i >= 1023) return;
  int delta = i - 511;
  const int mid = BUCK / 2; // 128
  float rp  = (float)delta;
  float sgn = (delta > 0) ? 1.f : ((delta < 0) ? -1.f : 0.f);
  float abs_pos = (delta < mid && delta > -mid) ? (float)(mid - 1) : fabsf(rp);
  float log_pos = ceilf(logf(abs_pos / (float)mid) * (float)(mid - 1) /
                        logf((float)(MAXP - 1) / (float)mid)) + (float)mid;
  float bucketf = (abs_pos <= (float)mid) ? rp : log_pos * sgn;
  int b = (int)bucketf; // truncation like astype(int32)
  int c2p = b + BUCK;  c2p = c2p < 0 ? 0 : (c2p > 511 ? 511 : c2p);
  int p2c = -b + BUCK; p2c = p2c < 0 ? 0 : (p2c > 511 ? 511 : p2c);
  c2ptab[i] = c2p;
  p2ctab[i] = p2c;
}

// ---------------------------------------------------------------------------
// Generic bf16 WMMA GEMM:  C[M,N] = A[M,Kd] * Bt[N,Kd]^T (+ bias[n])
// A, Bt row-major bf16 (leading dim == Kd).  128x64 block tile, 8 waves
// (4 along M x 2 along N), each wave owns a 32x32 tile:
//   2 A-frags x 2 B-frags -> 4 v_wmma_f32_16x16x32_bf16 per 32-K step.
// Output modes:
//   0: bf16 [B,H,S,D]   (m -> b,s ; n -> h,d)            Q/K/posQ/posK
//   1: bf16 [B,H,D,S]   (V transposed for the PV GEMM)
//   2: f32  row-major   off = z*sOz + m*ldO + n          raw scores / d_out
//   3: bf16 row-major   off = z*sOz + m*ldO + n          attn_out
// M % 128 == 0, N % 64 == 0, Kd % 32 == 0 for every launch in this file.
// ---------------------------------------------------------------------------
__global__ __launch_bounds__(256)
void gemm_bf16_wmma(const __bf16* __restrict__ A, long sAz,
                    const __bf16* __restrict__ Bt, long sBz,
                    const float* __restrict__ bias,
                    int M, int N, int Kd,
                    int mode, void* __restrict__ outp, long sOz, int ldO) {
  __shared__ __align__(16) __bf16 sA[128 * LDP];
  __shared__ __align__(16) __bf16 sB[64 * LDP];

  const int tid  = threadIdx.x;
  const int lane = tid & 31;
  const int wid  = tid >> 5;     // 0..7
  const int wm   = wid & 3;      // 4 waves along M (32 rows each)
  const int wn   = wid >> 2;     // 2 waves along N (32 cols each)
  const int m0   = blockIdx.x * 128;
  const int n0   = blockIdx.y * 64;
  const int z    = blockIdx.z;

  const __bf16* Az = A  + (long)z * sAz;
  const __bf16* Bz = Bt + (long)z * sBz;

  // staging assignments: A tile 128x32 (32B/thread), B tile 64x32 (16B/thread)
  const int ar = tid >> 1;            // 0..127
  const int ac = (tid & 1) * 16;      // 0 or 16
  const int br = tid >> 2;            // 0..63
  const int bc = (tid & 3) * 8;       // 0,8,16,24

  v8f acc00 = {}, acc01 = {}, acc10 = {}, acc11 = {};

  const int half = lane >> 4;         // 0/1
  const int l16  = lane & 15;

  for (int k0 = 0; k0 < Kd; k0 += 32) {
    const long abase = (long)(m0 + ar) * Kd + k0 + ac;
    const long bbase = (long)(n0 + br) * Kd + k0 + bc;
    *(uint4*)&sA[ar * LDP + ac]     = *(const uint4*)&Az[abase];
    *(uint4*)&sA[ar * LDP + ac + 8] = *(const uint4*)&Az[abase + 8];
    *(uint4*)&sB[br * LDP + bc]     = *(const uint4*)&Bz[bbase];
    if (k0 + 32 < Kd) { // prefetch next K tile -> global_prefetch
      __builtin_prefetch(&Az[abase + 32], 0, 0);
      __builtin_prefetch(&Bz[bbase + 32], 0, 0);
    }
    __syncthreads();

    // A fragment (16x32 bf16, ISA 7.12.2): lane l -> row (l&15);
    // lanes 0-15 hold K 0..7 & 16..23, lanes 16-31 hold K 8..15 & 24..31.
    union { v16bf v; uint4 q[2]; } a0, a1, b0, b1;
    const int ar0 = (wm * 32 + l16) * LDP;
    const int ar1 = (wm * 32 + 16 + l16) * LDP;
    a0.q[0] = *(const uint4*)&sA[ar0 + half * 8];
    a0.q[1] = *(const uint4*)&sA[ar0 + 16 + half * 8];
    a1.q[0] = *(const uint4*)&sA[ar1 + half * 8];
    a1.q[1] = *(const uint4*)&sA[ar1 + 16 + half * 8];
    // B fragment (32x16): lane l -> column (l&15), same K order as A.
    const int bc0 = (wn * 32 + l16) * LDP;
    const int bc1 = (wn * 32 + 16 + l16) * LDP;
    b0.q[0] = *(const uint4*)&sB[bc0 + half * 8];
    b0.q[1] = *(const uint4*)&sB[bc0 + 16 + half * 8];
    b1.q[0] = *(const uint4*)&sB[bc1 + half * 8];
    b1.q[1] = *(const uint4*)&sB[bc1 + 16 + half * 8];

    acc00 = __builtin_amdgcn_wmma_f32_16x16x32_bf16(false, a0.v, false, b0.v,
                                                    (short)0, acc00, false, false);
    acc01 = __builtin_amdgcn_wmma_f32_16x16x32_bf16(false, a0.v, false, b1.v,
                                                    (short)0, acc01, false, false);
    acc10 = __builtin_amdgcn_wmma_f32_16x16x32_bf16(false, a1.v, false, b0.v,
                                                    (short)0, acc10, false, false);
    acc11 = __builtin_amdgcn_wmma_f32_16x16x32_bf16(false, a1.v, false, b1.v,
                                                    (short)0, acc11, false, false);
    __syncthreads();
  }

  // Epilogue: C/D layout -> lane l, vgpr r holds (M = r + 8*(l>>4), N = l&15).
  auto store_tile = [&](v8f acc, int a, int t) {
    const int gm_base = m0 + wm * 32 + a * 16 + 8 * half;
    const int gn = n0 + wn * 32 + t * 16 + l16;
    const float bvv = bias ? bias[gn] : 0.f;
#pragma unroll
    for (int r = 0; r < 8; ++r) {
      const int gm = gm_base + r;
      const float val = acc[r] + bvv;
      if (mode == 0) {
        int bb = gm >> 9, s = gm & 511, h = gn >> 6, d = gn & 63;
        long off = ((((long)bb * NH + h) * SEQ + s) * HD + d);
        ((__bf16*)outp)[off] = (__bf16)val;
      } else if (mode == 1) {
        int bb = gm >> 9, s = gm & 511, h = gn >> 6, d = gn & 63;
        long off = ((((long)bb * NH + h) * HD + d) * SEQ + s);
        ((__bf16*)outp)[off] = (__bf16)val;
      } else if (mode == 2) {
        ((float*)outp)[(long)z * sOz + (long)gm * ldO + gn] = val;
      } else {
        ((__bf16*)outp)[(long)z * sOz + (long)gm * ldO + gn] = (__bf16)val;
      }
    }
  };
  store_tile(acc00, 0, 0);
  store_tile(acc01, 0, 1);
  store_tile(acc10, 1, 0);
  store_tile(acc11, 1, 1);
}

// ---------------------------------------------------------------------------
// Per-(q,h) row: gather c2p/p2c, scale, softmax over 512 keys, emit bf16 probs.
// Raw tensors are per-batch and L2-resident on MI455X (3 x 16.8MB << 192MB).
// ---------------------------------------------------------------------------
__global__ __launch_bounds__(256)
void softmax_gather(const float* __restrict__ ccraw,
                    const float* __restrict__ c2praw,
                    const float* __restrict__ p2craw,
                    const int* __restrict__ c2ptab,
                    const int* __restrict__ p2ctab,
                    __bf16* __restrict__ probs) {
  __shared__ float red[256];
  __shared__ float rowbuf[SEQ];
  const int q = blockIdx.x, h = blockIdx.y, tid = threadIdx.x;
  const float* ccr  = ccraw  + ((long)h * SEQ + q) * SEQ;
  const float* c2pr = c2praw + ((long)h * SEQ + q) * SEQ;
  const float* p2cb = p2craw + (long)h * SEQ * SEQ;

  float lmax = -1e30f;
  for (int k = tid; k < SEQ; k += 256) {
    float v = ccr[k]
            + c2pr[c2ptab[q - k + 511]]
            + p2cb[(long)k * SEQ + p2ctab[k - q + 511]];
    v *= SCALE_F;
    rowbuf[k] = v;
    lmax = fmaxf(lmax, v);
  }
  red[tid] = lmax; __syncthreads();
  for (int s = 128; s > 0; s >>= 1) {
    if (tid < s) red[tid] = fmaxf(red[tid], red[tid + s]);
    __syncthreads();
  }
  const float rmax = red[0];
  __syncthreads();

  float lsum = 0.f;
  for (int k = tid; k < SEQ; k += 256) {
    float e = __expf(rowbuf[k] - rmax);
    rowbuf[k] = e;
    lsum += e;
  }
  red[tid] = lsum; __syncthreads();
  for (int s = 128; s > 0; s >>= 1) {
    if (tid < s) red[tid] += red[tid + s];
    __syncthreads();
  }
  const float inv = 1.f / red[0];
  __bf16* pr = probs + ((long)h * SEQ + q) * SEQ;
  for (int k = tid; k < SEQ; k += 256) pr[k] = (__bf16)(rowbuf[k] * inv);
}

// ---------------------------------------------------------------------------
extern "C" void kernel_launch(void* const* d_in, const int* in_sizes, int n_in,
                              void* d_out, int out_size, void* d_ws, size_t ws_size,
                              hipStream_t stream) {
  (void)in_sizes; (void)n_in; (void)out_size; (void)ws_size;
  const float* inputs = (const float*)d_in[0];
  const float* rel    = (const float*)d_in[1];
  const float* Wq = (const float*)d_in[2];
  const float* bq = (const float*)d_in[3];
  const float* Wk = (const float*)d_in[4];
  const float* bk = (const float*)d_in[5];
  const float* Wv = (const float*)d_in[6];
  const float* bv = (const float*)d_in[7];
  const float* Wo = (const float*)d_in[8];
  const float* bo = (const float*)d_in[9];
  float* out = (float*)d_out;

  // -------- workspace carve-up --------
  char* ws = (char*)d_ws;
  const size_t NTOK    = (size_t)NB * SEQ;           // 4096
  const size_t SZ_ACT  = NTOK * HID * 2;             // bf16 activations (8.4MB)
  const size_t SZ_W    = (size_t)HID * HID * 2;      // bf16 weight (2MB)
  const size_t SZ_RAW  = (size_t)NH * SEQ * SEQ * 4; // per-b f32 raw (16.8MB)
  const size_t SZ_PROB = (size_t)NH * SEQ * SEQ * 2; // per-b bf16 probs
  size_t off = 0;
  auto take = [&](size_t sz) { size_t o = off; off += (sz + 255) & ~(size_t)255; return o; };
  __bf16* Xbf    = (__bf16*)(ws + take(SZ_ACT));   // inputs bf16
  __bf16* Rbf    = (__bf16*)(ws + take(SZ_ACT));   // rel_embeddings bf16
  __bf16* WqT    = (__bf16*)(ws + take(SZ_W));
  __bf16* WkT    = (__bf16*)(ws + take(SZ_W));
  __bf16* WvT    = (__bf16*)(ws + take(SZ_W));
  __bf16* WoT    = (__bf16*)(ws + take(SZ_W));
  __bf16* Qb     = (__bf16*)(ws + take(SZ_ACT));   // [B,H,S,D]
  __bf16* Kb     = (__bf16*)(ws + take(SZ_ACT));   // [B,H,S,D]
  __bf16* pQb    = (__bf16*)(ws + take(SZ_ACT));   // [B,H,P,D]
  __bf16* pKb    = (__bf16*)(ws + take(SZ_ACT));   // [B,H,P,D]
  __bf16* Vt     = (__bf16*)(ws + take(SZ_ACT));   // [B,H,D,S]
  __bf16* attn   = (__bf16*)(ws + take(SZ_ACT));   // [B*S, HID] bf16
  float*  ccraw  = (float*) (ws + take(SZ_RAW));   // per-b [H,S,S]
  float*  c2praw = (float*) (ws + take(SZ_RAW));
  float*  p2craw = (float*) (ws + take(SZ_RAW));
  __bf16* probs  = (__bf16*)(ws + take(SZ_PROB));
  int*    c2ptab = (int*)   (ws + take(1023 * 4));
  int*    p2ctab = (int*)   (ws + take(1023 * 4));

  const int nAct = (int)(NTOK * HID);
  dim3 blk(256);

  // -------- precision conversion + weight transposes + index tables --------
  cvt_f32_bf16<<<(nAct + 255) / 256, blk, 0, stream>>>(inputs, Xbf, nAct);
  cvt_f32_bf16<<<(nAct + 255) / 256, blk, 0, stream>>>(rel,    Rbf, nAct);
  dim3 tgrid(HID / 16, HID / 16), tblk(16, 16);
  transpose_cvt<<<tgrid, tblk, 0, stream>>>(Wq, WqT);
  transpose_cvt<<<tgrid, tblk, 0, stream>>>(Wk, WkT);
  transpose_cvt<<<tgrid, tblk, 0, stream>>>(Wv, WvT);
  transpose_cvt<<<tgrid, tblk, 0, stream>>>(Wo, WoT);
  build_pos_tables<<<4, 256, 0, stream>>>(c2ptab, p2ctab);

  // -------- projections (WMMA GEMMs) --------
  dim3 gproj((int)NTOK / 128, HID / 64, 1);
  gemm_bf16_wmma<<<gproj, blk, 0, stream>>>(Xbf, 0, WqT, 0, bq, (int)NTOK, HID, HID, 0, Qb,  0, 0);
  gemm_bf16_wmma<<<gproj, blk, 0, stream>>>(Xbf, 0, WkT, 0, bk, (int)NTOK, HID, HID, 0, Kb,  0, 0);
  gemm_bf16_wmma<<<gproj, blk, 0, stream>>>(Xbf, 0, WvT, 0, bv, (int)NTOK, HID, HID, 1, Vt,  0, 0);
  gemm_bf16_wmma<<<gproj, blk, 0, stream>>>(Rbf, 0, WqT, 0, bq, (int)NTOK, HID, HID, 0, pQb, 0, 0);
  gemm_bf16_wmma<<<gproj, blk, 0, stream>>>(Rbf, 0, WkT, 0, bk, (int)NTOK, HID, HID, 0, pKb, 0, 0);

  // -------- attention, serialized over batch (scratch stays in L2) --------
  const long hstride = (long)SEQ * HD;       // per-head stride in Q/K/posQ/posK
  const long rawz    = (long)SEQ * SEQ;      // per-head stride in raw tensors
  dim3 gattn(SEQ / 128, SEQ / 64, NH);
  dim3 gpv(SEQ / 128, HD / 64, NH);
  dim3 gsm(SEQ, NH);
  for (int b = 0; b < NB; ++b) {
    const __bf16* Qb_b  = Qb  + (long)b * NH * SEQ * HD;
    const __bf16* Kb_b  = Kb  + (long)b * NH * SEQ * HD;
    const __bf16* pQb_b = pQb + (long)b * NH * SEQ * HD;
    const __bf16* pKb_b = pKb + (long)b * NH * SEQ * HD;
    const __bf16* Vt_b  = Vt  + (long)b * NH * HD * SEQ;
    // content-content, c2p-raw, p2c-raw: [512 x 512 x 64] per head
    gemm_bf16_wmma<<<gattn, blk, 0, stream>>>(Qb_b, hstride, Kb_b,  hstride, nullptr,
                                              SEQ, SEQ, HD, 2, ccraw,  rawz, SEQ);
    gemm_bf16_wmma<<<gattn, blk, 0, stream>>>(Qb_b, hstride, pKb_b, hstride, nullptr,
                                              SEQ, SEQ, HD, 2, c2praw, rawz, SEQ);
    gemm_bf16_wmma<<<gattn, blk, 0, stream>>>(Kb_b, hstride, pQb_b, hstride, nullptr,
                                              SEQ, SEQ, HD, 2, p2craw, rawz, SEQ);
    // gather + scale + softmax -> bf16 probs
    softmax_gather<<<gsm, blk, 0, stream>>>(ccraw, c2praw, p2craw, c2ptab, p2ctab, probs);
    // probs x V : [512 x 64 x 512] per head -> attn_out[b*S.., h*64+d]
    gemm_bf16_wmma<<<gpv, blk, 0, stream>>>(probs, rawz, Vt_b, (long)HD * SEQ, nullptr,
                                            SEQ, HD, SEQ, 3,
                                            attn + (long)b * SEQ * HID, 64, HID);
  }

  // -------- output projection -> f32 d_out --------
  dim3 gout((int)NTOK / 128, HID / 64, 1);
  gemm_bf16_wmma<<<gout, blk, 0, stream>>>(attn, 0, WoT, 0, bo,
                                           (int)NTOK, HID, HID, 2, out, 0, HID);
}
